// ModelCustom_11871289606790
// MI455X (gfx1250) — matrix operand order repaired and checked by reading, not compiled, not run on hardware
//
#include <hip/hip_runtime.h>
#include <hip/hip_bf16.h>

typedef __attribute__((ext_vector_type(16))) __bf16 v16bf;
typedef __attribute__((ext_vector_type(8)))  __bf16 v8bf;
typedef __attribute__((ext_vector_type(8)))  float  v8f;
typedef __attribute__((ext_vector_type(4)))  unsigned int u32x4;
typedef __attribute__((ext_vector_type(4)))  int i32x4;
typedef __attribute__((ext_vector_type(8)))  int i32x8;

constexpr int kB = 64;      // batch
constexpr int kD = 512;     // hidden
constexpr int kV = 10000;   // vocab
constexpr int kT = 151;     // sequence length
constexpr int GRU_BLOCKS = 16;

#if defined(__AMDGCN__) && __has_builtin(__builtin_amdgcn_tensor_load_to_lds) && \
    __has_builtin(__builtin_amdgcn_s_wait_tensorcnt)
#define CDNA5_HAS_TDM 1
#else
#define CDNA5_HAS_TDM 0
#endif

union ATile { v16bf v; v8bf h[2]; };

__device__ __forceinline__ v8f wmma_bf16(v16bf a, v16bf b, v8f c) {
  // D = A(16x32) * B(32x16) + C, f32 accumulate
  return __builtin_amdgcn_wmma_f32_16x16x32_bf16(false, a, false, b, (short)0, c,
                                                 false, false);
}

#if CDNA5_HAS_TDM
// Flat contiguous global->LDS DMA via the Tensor Data Mover.
// 1-D tile of n8 8-byte units; D# packed per CDNA5 ISA:
//   group0: count=1 | lds_addr | global_addr[56:0] | type=2
//   group1: data_size=8B | tensor_dim0 = tile_dim0 = n8 | dim0_stride = n8
//   groups 2/3: zero (tensor rank <= 2)
__device__ __forceinline__ void tdm_load_flat_to_lds(const void* gsrc, unsigned lds_off,
                                                     unsigned n8) {
  unsigned long long ga = (unsigned long long)gsrc;
  u32x4 g0;
  g0.x = 1u;                                                   // count=1 (valid user D#)
  g0.y = lds_off;                                              // lds_addr (bytes)
  g0.z = (unsigned)(ga & 0xFFFFFFFFull);                       // global_addr[31:0]
  g0.w = (unsigned)((ga >> 32) & 0x01FFFFFFull) | (2u << 30);  // addr[56:32] | type=2
  i32x8 g1;
  g1[0] = (int)(3u << 16);                                     // data_size = 8 bytes
  g1[1] = (int)((n8 & 0xFFFFu) << 16);                         // tensor_dim0[15:0]
  g1[2] = (int)((n8 >> 16) | (1u << 16));                      // tensor_dim0[31:16] | tensor_dim1=1
  g1[3] = (int)((n8 & 0xFFFFu) << 16);                         // tile_dim0 = n8
  g1[4] = 0;                                                   // tile_dim1/2 unused (1-D)
  g1[5] = (int)n8;                                             // tensor_dim0_stride[31:0]
  g1[6] = 0;
  g1[7] = 0;
  i32x4 z4 = {};
#if __clang_major__ >= 23
  i32x8 z8 = {};
  __builtin_amdgcn_tensor_load_to_lds(g0, g1, z4, z4, z8, 0);
#else
  __builtin_amdgcn_tensor_load_to_lds(g0, g1, z4, z4, 0);
#endif
}
#endif

// ---------------------------------------------------------------- helpers ----
__global__ void cvt_bf16_kernel(const float* __restrict__ s, __bf16* __restrict__ d, int n) {
  int i = blockIdx.x * blockDim.x + threadIdx.x;
  int stride = gridDim.x * blockDim.x;
  for (; i < n; i += stride) d[i] = (__bf16)s[i];
}

// h0 = img_feat @ W_feat^T + b_feat  (f32 compute, stored bf16); also zeroes
// the grid-barrier counter used by the GRU kernel.
__global__ void feat_proj_kernel(const float* __restrict__ img, const float* __restrict__ Wf,
                                 const float* __restrict__ bfeat, __bf16* __restrict__ h0bf,
                                 unsigned* __restrict__ cnt) {
  if (blockIdx.x == 0 && threadIdx.x == 0) *cnt = 0u;
  int idx = blockIdx.x * blockDim.x + threadIdx.x;   // 0..32767
  int b = idx >> 9, j = idx & 511;
  const float* xr = img + (size_t)b * kD;
  const float* wr = Wf + (size_t)j * kD;
  float acc = bfeat[j];
  for (int k = 0; k < kD; ++k) acc = fmaf(xr[k], wr[k], acc);
  h0bf[idx] = (__bf16)acc;
}

// seq[0] = broadcast embed[0] over batch
__global__ void x0_kernel(const float* __restrict__ embed, __bf16* __restrict__ seq) {
  int idx = blockIdx.x * blockDim.x + threadIdx.x;   // 0..32767 (= b*512 + j)
  seq[idx] = (__bf16)embed[idx & 511];
}

// ---------------------------------------------------------------- GRU --------
__device__ __forceinline__ void grid_barrier(volatile unsigned* cnt, unsigned target) {
  __syncthreads();
  if (threadIdx.x == 0) {
    __threadfence();
    atomicAdd((unsigned*)cnt, 1u);
    while (*cnt < target) __builtin_amdgcn_s_sleep(1);
    __threadfence();
  }
  __syncthreads();
}

// 16 blocks x 256 threads (8 waves). 128 waves total = one 16x16 output tile
// per wave per gate. seq[t] doubles as the GRU hidden state (x_t == h_{t-1}
// for t>=2 in this autoregressive scan); only step 1 reads h0 separately.
__global__ __launch_bounds__(256)
void gru_kernel(const __bf16* __restrict__ Wih, const __bf16* __restrict__ Whh,
                const float* __restrict__ bih, const float* __restrict__ bhh,
                const __bf16* __restrict__ h0bf, __bf16* seq, unsigned* cnt) {
  const int wave  = threadIdx.x >> 5;
  const int lane  = threadIdx.x & 31;
  const int lhalf = lane >> 4;     // 0 | 1
  const int lmod  = lane & 15;
  const int tile  = blockIdx.x * 8 + wave;   // 0..127
  const int mt    = tile >> 5;               // 0..3  (M tiles over batch 64)
  const int ct    = tile & 31;               // 0..31 (N tiles over hidden 512)
  const int ncol  = ct * 16 + lmod;          // this lane's output column in [0,512)

  const float bi_r = bih[0 * kD + ncol], bi_z = bih[1 * kD + ncol], bi_n = bih[2 * kD + ncol];
  const float bh_r = bhh[0 * kD + ncol], bh_z = bhh[1 * kD + ncol], bh_n = bhh[2 * kD + ncol];

  // B-matrix row pointers (row of W = output column), per cdna5 WMMA B layout
  const __bf16* wir = Wih + (size_t)(0 * kD + ncol) * kD + lhalf * 16;
  const __bf16* wiz = Wih + (size_t)(1 * kD + ncol) * kD + lhalf * 16;
  const __bf16* win = Wih + (size_t)(2 * kD + ncol) * kD + lhalf * 16;
  const __bf16* whr = Whh + (size_t)(0 * kD + ncol) * kD + lhalf * 16;
  const __bf16* whz = Whh + (size_t)(1 * kD + ncol) * kD + lhalf * 16;
  const __bf16* whn = Whh + (size_t)(2 * kD + ncol) * kD + lhalf * 16;

  const int mrow = mt * 16 + lmod;           // A row this lane loads

  for (int t = 1; t < kT; ++t) {
    const __bf16* xsrc = seq + (size_t)(t - 1) * kB * kD;
    const __bf16* hsrc = (t == 1) ? h0bf : xsrc;
    const __bf16* xrow = xsrc + (size_t)mrow * kD;
    const __bf16* hrow = hsrc + (size_t)mrow * kD;

    v8f ir = {}, iz = {}, inn = {}, hr = {}, hz = {}, hn = {};
    for (int k0 = 0; k0 < kD; k0 += 32) {
      const int ka = k0 + lhalf * 16;
      ATile ax, ah;
      ax.h[0] = *(const v8bf*)(xrow + ka);
      ax.h[1] = *(const v8bf*)(xrow + ka + 8);
      ah.h[0] = *(const v8bf*)(hrow + ka);
      ah.h[1] = *(const v8bf*)(hrow + ka + 8);
      ir  = wmma_bf16(ax.v, *(const v16bf*)(wir + k0), ir);
      iz  = wmma_bf16(ax.v, *(const v16bf*)(wiz + k0), iz);
      inn = wmma_bf16(ax.v, *(const v16bf*)(win + k0), inn);
      hr  = wmma_bf16(ah.v, *(const v16bf*)(whr + k0), hr);
      hz  = wmma_bf16(ah.v, *(const v16bf*)(whz + k0), hz);
      hn  = wmma_bf16(ah.v, *(const v16bf*)(whn + k0), hn);
    }

    __bf16* drow = seq + (size_t)t * kB * kD;
#pragma unroll
    for (int r = 0; r < 8; ++r) {
      int mm = mt * 16 + r + lhalf * 8;      // C/D layout: VGPR r -> row r (+8 for upper half)
      float ho = (float)hsrc[(size_t)mm * kD + ncol];
      float rg = 1.f / (1.f + __expf(-(ir[r] + bi_r + hr[r] + bh_r)));
      float zg = 1.f / (1.f + __expf(-(iz[r] + bi_z + hz[r] + bh_z)));
      float ng = tanhf(inn[r] + bi_n + rg * (hn[r] + bh_n));
      float hv = (1.f - zg) * ng + zg * ho;
      drow[(size_t)mm * kD + ncol] = (__bf16)hv;
    }
    grid_barrier(cnt, (unsigned)(t * GRU_BLOCKS));
  }
}

// ---------------------------------------------------------------- proj -------
// grid (125, 64): block = (b, 80 vocab cols), all 151 timesteps.
// LDS: Bstage 80x512 bf16 (80KB, DMA'd by the Tensor Data Mover) +
// Cstage 160x80 f32 (50KB) -> coalesced t-contiguous stores into [B, V, T].
__global__ __launch_bounds__(256)
void proj_kernel(const __bf16* __restrict__ seq, const __bf16* __restrict__ Wp,
                 const float* __restrict__ bp, float* __restrict__ out) {
  extern __shared__ char psm[];
  __bf16* Bst = (__bf16*)psm;                    // [80][512], LDS offset 0
  float*  Cst = (float*)(psm + 80 * kD * 2);     // [160][80]

  const int vb    = blockIdx.x * 80;
  const int b     = blockIdx.y;
  const int tid   = threadIdx.x;
  const int wave  = tid >> 5;
  const int lane  = tid & 31;
  const int lhalf = lane >> 4;
  const int lmod  = lane & 15;

#if CDNA5_HAS_TDM
  // Stage W_proj rows [vb, vb+80) (contiguous 80KB) with one TDM descriptor.
  // This kernel's only LDS allocation is the dynamic block -> Bst at offset 0.
  if (wave == 0) {
    tdm_load_flat_to_lds(Wp + (size_t)vb * kD, 0u, (unsigned)(80 * kD * 2 / 8));
    __builtin_amdgcn_s_wait_tensorcnt(0);
  }
#else
  for (int e = tid; e < 80 * (kD / 8); e += 256) {
    int row = e / (kD / 8), c8 = e % (kD / 8);
    *(v8bf*)(Bst + row * kD + c8 * 8) = *(const v8bf*)(Wp + (size_t)(vb + row) * kD + c8 * 8);
  }
#endif
  __syncthreads();

  float bias[5];
#pragma unroll
  for (int nt = 0; nt < 5; ++nt) bias[nt] = bp[vb + nt * 16 + lmod];

  for (int mt = wave; mt < 10; mt += 8) {        // 10 M-tiles cover t=0..159 (pad)
    int trow = mt * 16 + lmod;
    if (trow > kT - 1) trow = kT - 1;            // clamp pad rows
    const __bf16* arow = seq + (size_t)(trow * kB + b) * kD;
    v8f acc[5] = {};
    for (int k0 = 0; k0 < kD; k0 += 32) {
      ATile a;
      a.h[0] = *(const v8bf*)(arow + k0 + lhalf * 16);
      a.h[1] = *(const v8bf*)(arow + k0 + lhalf * 16 + 8);
#pragma unroll
      for (int nt = 0; nt < 5; ++nt) {           // reuse A across 5 N-tiles
        const __bf16* brow = Bst + (size_t)(nt * 16 + lmod) * kD + k0 + lhalf * 16;
        acc[nt] = wmma_bf16(a.v, *(const v16bf*)brow, acc[nt]);
      }
    }
#pragma unroll
    for (int nt = 0; nt < 5; ++nt)
#pragma unroll
      for (int r = 0; r < 8; ++r) {
        int tl = mt * 16 + r + lhalf * 8;
        Cst[tl * 80 + nt * 16 + lmod] = acc[nt][r] + bias[nt];
      }
  }
  __syncthreads();

  const size_t obase = (size_t)b * kV * kT;
  for (int e = tid; e < 80 * kT; e += 256) {     // contiguous over t -> coalesced
    int vl = e / kT, t = e % kT;
    out[obase + (size_t)(vb + vl) * kT + t] = Cst[t * 80 + vl];
  }
}

// ---------------------------------------------------------------- argmax -----
// softmax is monotone => argmax over logits. One block per b, one thread per t:
// consecutive threads read consecutive t (coalesced over the T-contiguous dim).
__global__ void argmax_kernel(const float* __restrict__ out, float* __restrict__ tok) {
  int b = blockIdx.x;
  int t = threadIdx.x;
  if (t >= kT) return;
  const float* base = out + (size_t)b * kV * kT + t;
  float best = base[0];
  int bi = 0;
  for (int v = 1; v < kV; ++v) {
    float x = base[(size_t)v * kT];
    if (x > best) { best = x; bi = v; }
  }
  tok[b * kT + t] = (float)bi;
}

// ---------------------------------------------------------------- launch -----
extern "C" void kernel_launch(void* const* d_in, const int* in_sizes, int n_in,
                              void* d_out, int out_size, void* d_ws, size_t ws_size,
                              hipStream_t stream) {
  const float* img   = (const float*)d_in[0];
  const float* Wfeat = (const float*)d_in[1];
  const float* bfeat = (const float*)d_in[2];
  const float* embed = (const float*)d_in[3];
  const float* Wih   = (const float*)d_in[4];
  const float* Whh   = (const float*)d_in[5];
  const float* bih   = (const float*)d_in[6];
  const float* bhh   = (const float*)d_in[7];
  const float* Wproj = (const float*)d_in[8];
  const float* bproj = (const float*)d_in[9];
  float* out = (float*)d_out;

  // workspace layout (bytes)
  char* ws = (char*)d_ws;
  __bf16* Wih_bf = (__bf16*)(ws);                          // 3D*D  = 786432 elems
  __bf16* Whh_bf = (__bf16*)(ws + 1572864);                // 3D*D
  __bf16* Wp_bf  = (__bf16*)(ws + 3145728);                // V*D   = 5,120,000 elems
  __bf16* seq_bf = (__bf16*)(ws + 13385728);               // T*B*D = 4,947,968 elems
  __bf16* h0_bf  = (__bf16*)(ws + 23281664);               // B*D
  unsigned* cnt  = (unsigned*)(ws + 23347200);             // grid barrier counter

  cvt_bf16_kernel<<<512, 256, 0, stream>>>(Wih,   Wih_bf, 3 * kD * kD);
  cvt_bf16_kernel<<<512, 256, 0, stream>>>(Whh,   Whh_bf, 3 * kD * kD);
  cvt_bf16_kernel<<<2048, 256, 0, stream>>>(Wproj, Wp_bf, kV * kD);
  feat_proj_kernel<<<128, 256, 0, stream>>>(img, Wfeat, bfeat, h0_bf, cnt);
  x0_kernel<<<128, 256, 0, stream>>>(embed, seq_bf);

  gru_kernel<<<GRU_BLOCKS, 256, 0, stream>>>(Wih_bf, Whh_bf, bih, bhh, h0_bf, seq_bf, cnt);

  proj_kernel<<<dim3(125, 64), 256, 80 * kD * 2 + 160 * 80 * 4, stream>>>(
      seq_bf, Wp_bf, bproj, out);

  argmax_kernel<<<kB, 160, 0, stream>>>(out, out + (size_t)kB * kV * kT);
}
